// CondCopy_47373489275005
// MI455X (gfx1250) — matrix operand order repaired and checked by reading, not compiled, not run on hardware
//
#include <hip/hip_runtime.h>
#include <cmath>

// ---------------------------------------------------------------------------
// Pointer-generator LM head for MI455X (gfx1250, wave32).
// Big GEMM (1600x512x50257) done twice with bf16 WMMA + f32 online softmax
// to avoid materializing 322MB of logits. Output written once, log'd in place.
// ---------------------------------------------------------------------------

typedef __bf16 bf16;
typedef __attribute__((ext_vector_type(16))) __bf16 v16bf;
typedef __attribute__((ext_vector_type(8)))  __bf16 v8bf;
typedef __attribute__((ext_vector_type(8)))  float  v8f;

constexpr int V_ = 50257;
constexpr int H_ = 512;
constexpr int L_ = 50;
constexpr int B_ = 32;
constexpr int N_ = B_ * L_;                     // 1600 rows (b*L + l)
constexpr int CHUNK   = 256;                    // columns per wave in logits passes
constexpr int NCHUNK  = (V_ + CHUNK - 1) / CHUNK;   // 197
constexpr int LDS_STRIDE = H_ + 8;              // 520 bf16 -> bank-conflict-free

union FragA { v16bf v; v8bf h[2]; };

// A fragment (16x32 bf16, M x K) from LDS tile. Per ISA 16-bit A layout:
// lane 0-15 : row M=lane, K {k0..k0+7} in V0-3, K {k0+16..k0+23} in V4-7
// lane 16-31: row M=lane-16, K {k0+8..k0+15} and {k0+24..k0+31}
__device__ inline v16bf load_a_frag_lds(const bf16* At, int lane, int k0) {
  const int half = (lane >> 4) & 1;
  const int m    = lane & 15;
  const bf16* p  = At + m * LDS_STRIDE + k0 + half * 8;
  FragA f;
  f.h[0] = *(const v8bf*)(p);
  f.h[1] = *(const v8bf*)(p + 16);
  return f.v;
}

// B fragment (32x16 bf16, K x N). Column n of B == row n of a row-major
// weight matrix W[out, H]. Same K order as A: lanes 0-15 hold K {k0..k0+7} and
// {k0+16..k0+23} for N=lane; lanes 16-31 hold {k0+8..k0+15} and {k0+24..k0+31}.
__device__ inline v16bf load_b_frag_gl(const bf16* W, int row, int lane, int k0) {
  const int half = (lane >> 4) & 1;
  const bf16* p  = W + (size_t)row * H_ + k0 + half * 8;
  FragA f;
  f.h[0] = *(const v8bf*)(p);
  f.h[1] = *(const v8bf*)(p + 16);
  return f.v;
}

__device__ inline v8f wmma_bf16(v16bf a, v16bf b, v8f c) {
  return __builtin_amdgcn_wmma_f32_16x16x32_bf16(false, a, false, b,
                                                 (short)0, c, false, false);
}

// Cooperative load of a 16-row x 512-K bf16 A tile into LDS (128 threads).
__device__ inline void load_a_tile(bf16* At, const bf16* A, int row0, int tid) {
  for (int c = tid; c < 16 * (H_ / 8); c += 128) {
    const int r = c >> 6;            // 64 8-elem chunks per row
    const int o = (c & 63) * 8;
    *(v8bf*)&At[r * LDS_STRIDE + o] = *(const v8bf*)&A[(size_t)(row0 + r) * H_ + o];
  }
}

// --------------------------- small prep kernels ----------------------------

__global__ void cvt_f32_bf16(const float* __restrict__ src, bf16* __restrict__ dst, size_t n) {
  for (size_t k = blockIdx.x * (size_t)blockDim.x + threadIdx.x; k < n;
       k += (size_t)gridDim.x * blockDim.x)
    dst[k] = (bf16)src[k];
}

// emb = table[ctx]; clip row norm to <= 1; emit bf16.  128 threads per row.
__global__ void gather_norm(const int* __restrict__ ctx, const float* __restrict__ tab,
                            bf16* __restrict__ embb) {
  const int n = blockIdx.x;
  const int t = threadIdx.x;
  const int w = ctx[n];
  const float4 v = ((const float4*)(tab + (size_t)w * H_))[t];
  float ss = v.x * v.x + v.y * v.y + v.z * v.z + v.w * v.w;
#pragma unroll
  for (int off = 16; off; off >>= 1) ss += __shfl_xor(ss, off, 32);
  __shared__ float red[4];
  if ((t & 31) == 0) red[t >> 5] = ss;
  __syncthreads();
  const float nrm = sqrtf(red[0] + red[1] + red[2] + red[3]);
  const float scl = fminf(1.f, 1.f / fmaxf(nrm, 1e-12f));
  bf16* d = embb + (size_t)n * H_ + t * 4;
  d[0] = (bf16)(v.x * scl); d[1] = (bf16)(v.y * scl);
  d[2] = (bf16)(v.z * scl); d[3] = (bf16)(v.w * scl);
}

// --------------------------- H x H WMMA GEMM -------------------------------
// C[n, o] = act( sum_h A[n,h] * W[o,h] + bias[o] )
__global__ __launch_bounds__(128) void gemm_h512(const bf16* __restrict__ A,
                                                 const bf16* __restrict__ W,
                                                 const float* __restrict__ bias,
                                                 int do_tanh,
                                                 float* __restrict__ Cf,
                                                 bf16* __restrict__ Cbf) {
  __shared__ __align__(16) bf16 At[16 * LDS_STRIDE];
  const int row0 = blockIdx.x * 16;
  const int tid  = threadIdx.x;
  load_a_tile(At, A, row0, tid);
  __syncthreads();
  const int wave = tid >> 5, lane = tid & 31;
  const int half = lane >> 4, ln = lane & 15;
  for (int s = 0; s < 8; ++s) {                // 4 waves x 8 tiles = 32 col tiles
    const int col = (wave * 8 + s) * 16 + ln;
    v8f acc = {};
#pragma unroll
    for (int ks = 0; ks < H_ / 32; ++ks) {
      const v16bf a = load_a_frag_lds(At, lane, ks * 32);
      const v16bf b = load_b_frag_gl(W, col, lane, ks * 32);
      acc = wmma_bf16(a, b, acc);
    }
    const float bo = bias ? bias[col] : 0.f;
#pragma unroll
    for (int r = 0; r < 8; ++r) {
      float x = acc[r] + bo;
      if (do_tanh) x = tanhf(x);
      const size_t idx = (size_t)(row0 + half * 8 + r) * H_ + col;
      Cf[idx] = x;
      if (Cbf) Cbf[idx] = (bf16)x;
    }
  }
}

// switch = sigmoid(cvecs . W_copy + b_copy); gen_score = switch * sum(q)
__global__ void switch_gen(const float* __restrict__ cvecs, const float* __restrict__ qf,
                           const float* __restrict__ Wc, const float* __restrict__ bc,
                           float* __restrict__ gen) {
  const int n = blockIdx.x, t = threadIdx.x;   // 64 threads
  float d = 0.f, sq = 0.f;
  for (int k = t; k < H_; k += 64) {
    d  += cvecs[(size_t)n * H_ + k] * Wc[k];
    sq += qf[(size_t)n * H_ + k];
  }
#pragma unroll
  for (int off = 16; off; off >>= 1) {
    d += __shfl_xor(d, off, 32);
    sq += __shfl_xor(sq, off, 32);
  }
  __shared__ float rd[2], rs[2];
  if ((t & 31) == 0) { rd[t >> 5] = d; rs[t >> 5] = sq; }
  __syncthreads();
  if (t == 0) {
    const float dot = rd[0] + rd[1] + bc[0];
    const float sw  = 1.f / (1.f + __expf(-dot));
    gen[n] = sw * (rs[0] + rs[1]);
  }
}

// Causal attention softmax over {scores[0..i], gen_score}.  64 threads/row.
__global__ void attn_softmax(const float* __restrict__ qf, const float* __restrict__ cvecs,
                             const float* __restrict__ gen,
                             float* __restrict__ aPtr, float* __restrict__ aGen) {
  const int n  = blockIdx.x;
  const int bb = n / L_, ii = n - bb * L_;
  const int t  = threadIdx.x;
  __shared__ __align__(16) float qs[H_];
  __shared__ float sc[L_];
  __shared__ float inv_s;
  for (int k = t; k < H_; k += 64) qs[k] = qf[(size_t)n * H_ + k];
  __syncthreads();
  if (t <= ii) {
    const float* c = cvecs + (size_t)(bb * L_ + t) * H_;
    float d = 0.f;
    for (int k = 0; k < H_; k += 4)
      d += qs[k] * c[k] + qs[k + 1] * c[k + 1] + qs[k + 2] * c[k + 2] + qs[k + 3] * c[k + 3];
    sc[t] = d;
  }
  __syncthreads();
  if (t == 0) {
    float mx = gen[n];
    for (int j = 0; j <= ii; ++j) mx = fmaxf(mx, sc[j]);
    const float eg = __expf(gen[n] - mx);
    float s = eg;
    for (int j = 0; j <= ii; ++j) { const float e = __expf(sc[j] - mx); sc[j] = e; s += e; }
    inv_s = 1.f / s;
    aGen[n] = eg * inv_s;
  }
  __syncthreads();
  if (t < L_) aPtr[(size_t)n * L_ + t] = (t <= ii) ? sc[t] * inv_s : 0.f;
}

// ------------------- logits pass 1: per-chunk online max/sum ----------------
__global__ __launch_bounds__(128) void logits_pass1(const bf16* __restrict__ A,
                                                    const bf16* __restrict__ W,
                                                    const float* __restrict__ bout,
                                                    float* __restrict__ cM,
                                                    float* __restrict__ cS) {
  __shared__ __align__(16) bf16 At[16 * LDS_STRIDE];
  const int row0 = blockIdx.x * 16;
  const int tid  = threadIdx.x;
  load_a_tile(At, A, row0, tid);
  __syncthreads();
  const int wave = tid >> 5, lane = tid & 31;
  const int chunk = blockIdx.y * 4 + wave;
  if (chunk >= NCHUNK) return;
  const int half = lane >> 4, ln = lane & 15;
  float M[8], S[8];
#pragma unroll
  for (int r = 0; r < 8; ++r) { M[r] = -INFINITY; S[r] = 0.f; }
  for (int ct = 0; ct < CHUNK / 16; ++ct) {
    const int col  = chunk * CHUNK + ct * 16 + ln;
    const int colc = col < V_ ? col : V_ - 1;
    v8f acc = {};
#pragma unroll
    for (int ks = 0; ks < H_ / 32; ++ks) {
      const v16bf a = load_a_frag_lds(At, lane, ks * 32);
      const v16bf b = load_b_frag_gl(W, colc, lane, ks * 32);
      acc = wmma_bf16(a, b, acc);
    }
    if (col < V_) {
      const float bo = bout[col];
#pragma unroll
      for (int r = 0; r < 8; ++r) {
        const float x  = acc[r] + bo;
        const float m2 = fmaxf(M[r], x);
        S[r] = S[r] * __expf(M[r] - m2) + __expf(x - m2);
        M[r] = m2;
      }
    }
  }
  // merge the 16 column-lanes of each half (rows are per-half)
#pragma unroll
  for (int off = 1; off < 16; off <<= 1) {
#pragma unroll
    for (int r = 0; r < 8; ++r) {
      const float Mo = __shfl_xor(M[r], off, 32);
      const float So = __shfl_xor(S[r], off, 32);
      const float m2 = fmaxf(M[r], Mo);
      S[r] = S[r] * __expf(M[r] - m2) + So * __expf(Mo - m2);
      M[r] = m2;
    }
  }
  if (ln == 0) {
#pragma unroll
    for (int r = 0; r < 8; ++r) {
      const int n = row0 + half * 8 + r;
      cM[(size_t)n * NCHUNK + chunk] = M[r];
      cS[(size_t)n * NCHUNK + chunk] = S[r];
    }
  }
}

__global__ void merge_ms(const float* __restrict__ cM, const float* __restrict__ cS,
                         float* __restrict__ rM, float* __restrict__ rS) {
  const int n = blockIdx.x * blockDim.x + threadIdx.x;
  if (n >= N_) return;
  float M = -INFINITY, S = 0.f;
  for (int c = 0; c < NCHUNK; ++c) {
    const float m  = cM[(size_t)n * NCHUNK + c];
    const float s  = cS[(size_t)n * NCHUNK + c];
    const float m2 = fmaxf(M, m);
    S = S * __expf(M - m2) + s * __expf(m - m2);
    M = m2;
  }
  rM[n] = M; rS[n] = S;
}

// ---- logits pass 2: recompute, write p0 = softmax * a_gen (permuted) -------
__global__ __launch_bounds__(128) void logits_pass2(const bf16* __restrict__ A,
                                                    const bf16* __restrict__ W,
                                                    const float* __restrict__ bout,
                                                    const float* __restrict__ rM,
                                                    const float* __restrict__ rS,
                                                    const float* __restrict__ aGen,
                                                    float* __restrict__ out) {
  __shared__ __align__(16) bf16 At[16 * LDS_STRIDE];
  const int row0 = blockIdx.x * 16;
  const int tid  = threadIdx.x;
  load_a_tile(At, A, row0, tid);
  __syncthreads();
  const int wave = tid >> 5, lane = tid & 31;
  const int chunk = blockIdx.y * 4 + wave;
  if (chunk >= NCHUNK) return;
  const int half = lane >> 4, ln = lane & 15;
  float  Mr[8], scl[8];
  size_t obase[8];
#pragma unroll
  for (int r = 0; r < 8; ++r) {
    const int n  = row0 + half * 8 + r;
    const int bb = n / L_;
    const int ii = n - bb * L_;
    obase[r] = (size_t)(ii * B_ + bb) * V_;   // output row = l*B + b
    Mr[r]  = rM[n];
    scl[r] = aGen[n] / rS[n];
  }
  for (int ct = 0; ct < CHUNK / 16; ++ct) {
    const int col  = chunk * CHUNK + ct * 16 + ln;
    const int colc = col < V_ ? col : V_ - 1;
    v8f acc = {};
#pragma unroll
    for (int ks = 0; ks < H_ / 32; ++ks) {
      const v16bf a = load_a_frag_lds(At, lane, ks * 32);
      const v16bf b = load_b_frag_gl(W, colc, lane, ks * 32);
      acc = wmma_bf16(a, b, acc);
    }
    if (col < V_) {
      const float bo = bout[col];
#pragma unroll
      for (int r = 0; r < 8; ++r)
        out[obase[r] + col] = __expf(acc[r] + bo - Mr[r]) * scl[r];
    }
  }
}

// Scatter pointer probabilities: p[l*B+b, ctx[b,j]] += a_ptr[b,l,j], j<=l.
__global__ void scatter_ptr(const int* __restrict__ ctx, const float* __restrict__ aPtr,
                            float* __restrict__ out) {
  const int n  = blockIdx.x;
  const int bb = n / L_, ii = n - bb * L_;
  const int j  = threadIdx.x;                  // 64 threads
  if (j <= ii && j < L_) {
    const int w = ctx[bb * L_ + j];
    atomicAdd(out + (size_t)(ii * B_ + bb) * V_ + w, aPtr[(size_t)n * L_ + j]);
  }
}

__global__ void log_inplace(float* __restrict__ out, size_t n) {
  for (size_t k = blockIdx.x * (size_t)blockDim.x + threadIdx.x; k < n;
       k += (size_t)gridDim.x * blockDim.x)
    out[k] = __logf(out[k]);
}

// ---------------------------------------------------------------------------

extern "C" void kernel_launch(void* const* d_in, const int* in_sizes, int n_in,
                              void* d_out, int out_size, void* d_ws, size_t ws_size,
                              hipStream_t stream) {
  (void)in_sizes; (void)n_in; (void)out_size; (void)ws_size;
  const int*   ctx   = (const int*)  d_in[0];
  const float* table = (const float*)d_in[1];
  const float* Wctx2 = (const float*)d_in[2];
  const float* Wloc  = (const float*)d_in[3];
  const float* bloc  = (const float*)d_in[4];
  const float* Wout  = (const float*)d_in[5];
  const float* bout  = (const float*)d_in[6];
  const float* Wcopy = (const float*)d_in[7];
  const float* bcopy = (const float*)d_in[8];
  float* out = (float*)d_out;

  char* ws = (char*)d_ws;
  size_t off = 0;
  auto take = [&](size_t bytes) -> char* {
    char* p = ws + off;
    off = (off + bytes + 255) & ~(size_t)255;
    return p;
  };
  bf16*  wctx2b = (bf16*) take((size_t)H_ * H_ * sizeof(bf16));
  bf16*  wlocb  = (bf16*) take((size_t)H_ * H_ * sizeof(bf16));
  bf16*  woutb  = (bf16*) take((size_t)V_ * H_ * sizeof(bf16));
  bf16*  embb   = (bf16*) take((size_t)N_ * H_ * sizeof(bf16));
  float* cvecsF = (float*)take((size_t)N_ * H_ * sizeof(float));
  bf16*  cvecsB = (bf16*) take((size_t)N_ * H_ * sizeof(bf16));
  float* qF     = (float*)take((size_t)N_ * H_ * sizeof(float));
  float* genS   = (float*)take((size_t)N_ * sizeof(float));
  float* aPtr   = (float*)take((size_t)N_ * L_ * sizeof(float));
  float* aGen   = (float*)take((size_t)N_ * sizeof(float));
  float* chunkM = (float*)take((size_t)N_ * NCHUNK * sizeof(float));
  float* chunkS = (float*)take((size_t)N_ * NCHUNK * sizeof(float));
  float* rowM   = (float*)take((size_t)N_ * sizeof(float));
  float* rowS   = (float*)take((size_t)N_ * sizeof(float));

  // 1) weight conversion + embedding gather/normalize
  cvt_f32_bf16<<<256, 256, 0, stream>>>(Wctx2, wctx2b, (size_t)H_ * H_);
  cvt_f32_bf16<<<256, 256, 0, stream>>>(Wloc,  wlocb,  (size_t)H_ * H_);
  cvt_f32_bf16<<<2048, 256, 0, stream>>>(Wout, woutb, (size_t)V_ * H_);
  gather_norm<<<N_, 128, 0, stream>>>(ctx, table, embb);

  // 2) cvecs = emb @ Wctx2^T ; q = tanh(cvecs @ Wloc^T + b_loc)
  gemm_h512<<<N_ / 16, 128, 0, stream>>>(embb,   wctx2b, nullptr, 0, cvecsF, cvecsB);
  gemm_h512<<<N_ / 16, 128, 0, stream>>>(cvecsB, wlocb,  bloc,    1, qF, nullptr);

  // 3) gen_score + causal attention softmax
  switch_gen<<<N_, 64, 0, stream>>>(cvecsF, qF, Wcopy, bcopy, genS);
  attn_softmax<<<N_, 64, 0, stream>>>(qF, cvecsF, genS, aPtr, aGen);

  // 4) vocab softmax: pass1 (chunk max/sum) -> merge -> pass2 (write probs)
  dim3 g1(N_ / 16, (NCHUNK + 3) / 4);
  logits_pass1<<<g1, 128, 0, stream>>>(cvecsB, woutb, bout, chunkM, chunkS);
  merge_ms<<<(N_ + 255) / 256, 256, 0, stream>>>(chunkM, chunkS, rowM, rowS);
  logits_pass2<<<g1, 128, 0, stream>>>(cvecsB, woutb, bout, rowM, rowS, aGen, out);

  // 5) pointer scatter-add + log
  scatter_ptr<<<N_, 64, 0, stream>>>(ctx, aPtr, out);
  log_inplace<<<4096, 256, 0, stream>>>(out, (size_t)N_ * V_);
}
